// MultiheadTemporalLatentAttentionHF_75960791597102
// MI455X (gfx1250) — compile-verified, matrix operand order checked by reading, not run on hardware
//
#include <hip/hip_runtime.h>
#include <hip/hip_bf16.h>
#include <stdint.h>

// ---------------------------------------------------------------------------
// MultiheadTemporalLatentAttention for MI455X (gfx1250, wave32, WMMA).
// Heavy contractions: v_wmma_f32_16x16x32_bf16 (f32 accumulate), operands
// bf16 end-to-end. GEMM uses ping-pong LDS with GLOBAL_LOAD_ASYNC_TO_LDS_B128
// (ASYNCcnt) to overlap staging with matrix math. The temporal gate Wg is
// 2x2 block-diagonal (chunk mask, DOWN=2) -> <=2 gated rows per kv_t row.
// ---------------------------------------------------------------------------

#define B_    2
#define S_    2048
#define E_    2048
#define H_    16
#define NOPE_ 64
#define ROPE_ 32
#define V_    64
#define KVR_  256
#define QKD_  96          // NOPE + ROPE
#define KVCAT_ 288        // KVR + ROPE  (latent || rope, fused score dim)
#define SCALE_ 0.10206207261596577f   // 96^-0.5

typedef __attribute__((ext_vector_type(16))) __bf16 v16bf;
typedef __attribute__((ext_vector_type(8)))  float  v8f;

union Frag16 { v16bf v; uint4 u4[2]; };
union U8h    { uint4 q; unsigned short h[8]; };

__device__ __forceinline__ unsigned short f32_bf16(float f) {
  unsigned int u = __float_as_uint(f);
  u += 0x7fffu + ((u >> 16) & 1u);          // round-to-nearest-even
  return (unsigned short)(u >> 16);
}
__device__ __forceinline__ float bf16_f32(unsigned short h) {
  return __uint_as_float((unsigned int)h << 16);
}

__device__ __forceinline__ v8f wmma_bf16(const Frag16 &a, const Frag16 &b, v8f c) {
  return __builtin_amdgcn_wmma_f32_16x16x32_bf16(false, a.v, false, b.v,
                                                 (short)0, c, false, false);
}
__device__ __forceinline__ v8f v8zero() {
  v8f z = {0.f, 0.f, 0.f, 0.f, 0.f, 0.f, 0.f, 0.f};
  return z;
}

// -------- CDNA5 async global->LDS copy (16B/lane, ASYNCcnt tracked) --------
// Generic LDS pointers carry the LDS byte offset in their low 32 bits
// (ISA §10.2 aperture mapping), so truncation gives the ds address operand.
__device__ __forceinline__ unsigned lds_off(const void* p) {
  return (unsigned)(uintptr_t)p;
}
__device__ __forceinline__ void async_copy16(unsigned dst, unsigned long long src) {
  asm volatile("global_load_async_to_lds_b128 %0, %1, off"
               :: "v"(dst), "v"(src) : "memory");
}
__device__ __forceinline__ void wait_async0() {
  asm volatile("s_wait_asynccnt 0x0" ::: "memory");
}

// ---------------------------------------------------------------------------
// f32 -> bf16 bulk conversion, 4 elements / thread-iteration (b128 loads)
// ---------------------------------------------------------------------------
__global__ __launch_bounds__(256)
void cvt_f32_bf16_v4(const float4* __restrict__ in, uint2* __restrict__ out, long n4)
{
  long i = (long)blockIdx.x * blockDim.x + threadIdx.x;
  const long stride = (long)gridDim.x * blockDim.x;
  for (; i < n4; i += stride) {
    float4 f = in[i];
    uint2 o;
    o.x = (unsigned)f32_bf16(f.x) | ((unsigned)f32_bf16(f.y) << 16);
    o.y = (unsigned)f32_bf16(f.z) | ((unsigned)f32_bf16(f.w) << 16);
    out[i] = o;
  }
}

// ---------------------------------------------------------------------------
// Generic batched GEMM:  C[M,N] = A[M,K] * B   (bf16 in, f32 accumulate)
//   BNK==1 : B stored (N,K) row-major ("A @ W^T"), async-staged
//   BNK==0 : B stored (K,N) row-major ("A @ B"), register transpose (N%8==0)
//   OB16   : store C as bf16 (feeds another WMMA stage) else f32
// Tile 128(M) x 64(N) x 32(K); 8 waves, each owns a 32x32 patch -> 4 WMMAs
// per K-step. Ping-pong LDS: async-copy tile k+1 while WMMAs run on tile k;
// one s_wait_asynccnt + one barrier per step.
// Grid: (ceil(N/64), M/128, batch).  M%128==0, K%32==0 assumed.
// ---------------------------------------------------------------------------
#define GBM 128
#define GBN 64
#define GBK 32

template<int BNK, int OB16>
__global__ __launch_bounds__(256)
void gemm_kern(const unsigned short* __restrict__ A, int lda, long sA,
               const unsigned short* __restrict__ Bm, int ldb, long sB,
               void* __restrict__ Cv, int ldc, long sC,
               int N, int K)
{
  __shared__ __align__(16) unsigned short lA[2][GBM * GBK];   // 2 x 8 KB
  __shared__ __align__(16) unsigned short lB[2][GBN * GBK];   // 2 x 4 KB

  const int bat = blockIdx.z;
  const unsigned short* Ab = A  + (long)bat * sA;
  const unsigned short* Bb = Bm + (long)bat * sB;

  const int n0 = blockIdx.x * GBN;
  const int m0 = blockIdx.y * GBM;
  const int tid = threadIdx.x;
  const int wid = tid >> 5, lane = tid & 31;
  const int lane16 = lane & 15, half = lane >> 4;
  const int wm = (wid >> 1) * 32;          // wave M offset in tile (4 waves)
  const int wn = (wid & 1) * 32;           // wave N offset in tile (2 waves)

  // ---- fixed per-thread staging addresses ----
  const int ar0 = tid >> 2,        ac0 = tid & 3;             // A rows 0..63
  const int ar1 = (tid + 256) >> 2;                           // A rows 64..127
  const uint4* aSrc0 = (const uint4*)(Ab + (long)(m0 + ar0) * lda) + ac0;
  const uint4* aSrc1 = (const uint4*)(Ab + (long)(m0 + ar1) * lda) + ac0;
  unsigned aOff0[2], aOff1[2], bOff[2];
  aOff0[0] = lds_off(&lA[0][tid * 8]);         aOff0[1] = lds_off(&lA[1][tid * 8]);
  aOff1[0] = lds_off(&lA[0][(tid + 256) * 8]); aOff1[1] = lds_off(&lA[1][(tid + 256) * 8]);
  bOff[0]  = lds_off(&lB[0][tid * 8]);         bOff[1]  = lds_off(&lB[1][tid * 8]);

  // NT: one uint4, row n = tid>>2 (async). NN: one uint4 along n, transposed.
  const int bnRow = tid >> 2;
  const bool bnOk = (n0 + bnRow) < N;
  const uint4* bSrcNT = (const uint4*)(Bb + (long)(n0 + bnRow) * ldb) + (tid & 3);
  const int nnK = tid >> 3, nnN8 = tid & 7;
  const bool nnOk = (n0 + nnN8 * 8) < N;
  const unsigned short* bSrcNN = Bb + (long)nnK * ldb + n0 + nnN8 * 8;
  const int nnBase = nnN8 * 8 * GBK + nnK;
  const uint4 zq = {0u, 0u, 0u, 0u};
  if (BNK && !bnOk) {           // masked-N columns: keep both buffers zero
    *(uint4*)&lB[0][tid * 8] = zq;
    *(uint4*)&lB[1][tid * 8] = zq;
  }

  // ---- fixed fragment read addresses (per buffer) ----
  const uint4 *aF0[2], *aF1[2], *bF0[2], *bF1[2];
#pragma unroll
  for (int p = 0; p < 2; ++p) {
    aF0[p] = (const uint4*)(lA[p] + (wm      + lane16) * GBK);
    aF1[p] = (const uint4*)(lA[p] + (wm + 16 + lane16) * GBK);
    bF0[p] = (const uint4*)(lB[p] + (wn      + lane16) * GBK);
    bF1[p] = (const uint4*)(lB[p] + (wn + 16 + lane16) * GBK);
  }

  v8f a00 = v8zero(), a01 = v8zero(), a10 = v8zero(), a11 = v8zero();

  auto stage = [&](int p) {
    async_copy16(aOff0[p], (unsigned long long)(uintptr_t)aSrc0);
    async_copy16(aOff1[p], (unsigned long long)(uintptr_t)aSrc1);
    aSrc0 += 4; aSrc1 += 4;
    if (BNK) {
      if (bnOk) async_copy16(bOff[p], (unsigned long long)(uintptr_t)bSrcNT);
      bSrcNT += 4;
    } else {
      U8h u; u.q = nnOk ? *(const uint4*)bSrcNN : zq;
#pragma unroll
      for (int j = 0; j < 8; ++j) lB[p][nnBase + j * GBK] = u.h[j];
      bSrcNN += (long)GBK * ldb;
    }
    __builtin_prefetch(aSrc0, 0, 1);        // global_prefetch: tile k+2
  };

  auto compute = [&](int p) {
    Frag16 fa0, fa1, fb0, fb1;
    fa0.u4[0] = aF0[p][half]; fa0.u4[1] = aF0[p][2 + half];
    fa1.u4[0] = aF1[p][half]; fa1.u4[1] = aF1[p][2 + half];
    fb0.u4[0] = bF0[p][half]; fb0.u4[1] = bF0[p][2 + half];
    fb1.u4[0] = bF1[p][half]; fb1.u4[1] = bF1[p][2 + half];
    a00 = wmma_bf16(fa0, fb0, a00);
    a01 = wmma_bf16(fa0, fb1, a01);
    a10 = wmma_bf16(fa1, fb0, a10);
    a11 = wmma_bf16(fa1, fb1, a11);
  };

  stage(0);
  wait_async0();
  __syncthreads();
  int p = 0;
  for (int k0 = GBK; k0 < K; k0 += GBK) {
    stage(1 - p);              // async copies overlap the WMMAs below
    compute(p);
    wait_async0();
    __syncthreads();
    p ^= 1;
  }
  compute(p);

  // ---- store (C layout: VGPR r -> row r+8*half, lane16 -> col) ----
  v8f accs[2][2] = {{a00, a01}, {a10, a11}};
#pragma unroll
  for (int mi = 0; mi < 2; ++mi)
#pragma unroll
    for (int ni = 0; ni < 2; ++ni) {
      const int col = n0 + wn + ni * 16 + lane16;
      if (col >= N) continue;
      const long rowb = (long)(m0 + wm + mi * 16 + 8 * half);
      if (OB16) {
        unsigned short* Cb = (unsigned short*)Cv + (long)bat * sC + rowb * ldc + col;
#pragma unroll
        for (int r = 0; r < 8; ++r) Cb[(long)r * ldc] = f32_bf16(accs[mi][ni][r]);
      } else {
        float* Cb = (float*)Cv + (long)bat * sC + rowb * ldc + col;
#pragma unroll
        for (int r = 0; r < 8; ++r) Cb[(long)r * ldc] = accs[mi][ni][r];
      }
    }
}

// ---------------------------------------------------------------------------
// LayerNorm over kv_c (256) + RoPE on k_pe (32) -> kv_cat f32 (B,S,288)
// ---------------------------------------------------------------------------
__global__ __launch_bounds__(256)
void ln_rope_kernel(const float* __restrict__ kv, const float* __restrict__ g,
                    const float* __restrict__ bb, float* __restrict__ kv_cat)
{
  const long row = blockIdx.x;                 // b*S + s
  const int  s   = (int)(row % S_);
  const int  tid = threadIdx.x;
  const float* src = kv + row * KVCAT_;
  float x = src[tid];

  float s1 = x, s2 = x * x;
#pragma unroll
  for (int off = 1; off < 32; off <<= 1) {
    s1 += __shfl_xor(s1, off, 32);
    s2 += __shfl_xor(s2, off, 32);
  }
  __shared__ float r1[8], r2[8];
  if ((tid & 31) == 0) { r1[tid >> 5] = s1; r2[tid >> 5] = s2; }
  __syncthreads();
  float ts1 = 0.f, ts2 = 0.f;
#pragma unroll
  for (int i = 0; i < 8; ++i) { ts1 += r1[i]; ts2 += r2[i]; }
  const float mean = ts1 * (1.f / 256.f);
  const float var  = ts2 * (1.f / 256.f) - mean * mean;
  const float inv  = rsqrtf(var + 1e-5f);

  float* dst = kv_cat + row * KVCAT_;
  dst[tid] = (x - mean) * inv * g[tid] + bb[tid];

  if (tid < 16) {                              // RoPE on the 32 pe dims
    const int j = tid;
    const float freq = __expf(-(float)j * 0.5756462732485115f); // ln(1e4)/16
    float sn, cs;
    __sincosf((float)s * freq, &sn, &cs);
    const float x0 = src[KVR_ + 2 * j], x1 = src[KVR_ + 2 * j + 1];
    dst[KVR_ + 2 * j]     = x0 * cs - x1 * sn;
    dst[KVR_ + 2 * j + 1] = x0 * sn + x1 * cs;
  }
}

// ---------------------------------------------------------------------------
// RoPE on q_pe (bf16 in) -> qh[..., 256:288] bf16   (qh is (B,S,H,288))
// ---------------------------------------------------------------------------
__global__ __launch_bounds__(256)
void rope_q_kernel(const unsigned short* __restrict__ qb, unsigned short* __restrict__ qh)
{
  long idx = (long)blockIdx.x * blockDim.x + threadIdx.x;   // B*S*H*16
  const int  j  = (int)(idx & 15);
  const int  h  = (int)((idx >> 4) & 15);
  const long bs = idx >> 8;
  const int  s  = (int)(bs % S_);
  const unsigned short* qp = qb + bs * (H_ * QKD_) + h * QKD_ + NOPE_ + 2 * j;
  const float freq = __expf(-(float)j * 0.5756462732485115f);
  float sn, cs;
  __sincosf((float)s * freq, &sn, &cs);
  const float x0 = bf16_f32(qp[0]), x1 = bf16_f32(qp[1]);
  unsigned short* o = qh + (bs * H_ + h) * KVCAT_ + KVR_ + 2 * j;
  o[0] = f32_bf16(x0 * cs - x1 * sn);
  o[1] = f32_bf16(x0 * sn + x1 * cs);
}

// ---------------------------------------------------------------------------
// Analytic positional encodings through fc_c / fc_p:
//   C2[t] = pos_enc_row(t>>1) @ fc_c_w^T + fc_c_b
//   P2[t] = pos_enc_row(t)    @ fc_p_w^T + fc_p_b
// ---------------------------------------------------------------------------
__global__ __launch_bounds__(64)
void pos_fc_kernel(const float* __restrict__ wc, const float* __restrict__ bc,
                   const float* __restrict__ wp, const float* __restrict__ bp,
                   float* __restrict__ C2, float* __restrict__ P2)
{
  const int t = blockIdx.x, o = threadIdx.x;
  const float* wrc = wc + o * KVR_;
  const float* wrp = wp + o * KVR_;
  const float pc = (float)(t >> 1);
  const float pp = (float)t;
  float ac = 0.f, ap = 0.f;
  for (int j = 0; j < 128; ++j) {
    const float div = __expf(-(float)j * 0.07195578415606394f); // ln(1e4)/128
    float sn, cs;
    __sincosf(pc * div, &sn, &cs);
    ac += sn * wrc[2 * j] + cs * wrc[2 * j + 1];
    __sincosf(pp * div, &sn, &cs);
    ap += sn * wrp[2 * j] + cs * wrp[2 * j + 1];
  }
  C2[t * 64 + o] = ac + bc[o];
  P2[t * 64 + o] = ap + bp[o];
}

// ---------------------------------------------------------------------------
// kv_t_cat[b,t] via the 2x2 block-diagonal gate; emits bf16 in BOTH layouts:
//   kvh  (B,S,288)  for the score pass
//   kvhT (B,256,S)  for the P@V pass
// ---------------------------------------------------------------------------
__global__ __launch_bounds__(64)
void kv_gate_kernel(const float* __restrict__ C2, const float* __restrict__ P2,
                    const float* __restrict__ kv_cat,
                    unsigned short* __restrict__ kvh, unsigned short* __restrict__ kvhT)
{
  const long bt = blockIdx.x;
  const int  t  = (int)(bt % S_);
  const int  b  = (int)(bt / S_);
  const int  tid = threadIdx.x;
  const int  s0 = t & ~1;
  const bool odd = (t & 1) != 0;

  const float c2 = C2[(long)t * 64 + tid];
  float dS = c2 * P2[(long)t * 64 + tid];
  float dP = odd ? c2 * P2[(long)s0 * 64 + tid] : 0.f;
#pragma unroll
  for (int off = 1; off < 32; off <<= 1) {
    dS += __shfl_xor(dS, off, 32);
    dP += __shfl_xor(dP, off, 32);
  }
  __shared__ float red[4];
  if ((tid & 31) == 0) { red[(tid >> 5) * 2] = dS; red[(tid >> 5) * 2 + 1] = dP; }
  __syncthreads();
  const float wS = 1.f / (1.f + __expf(-(red[0] + red[2])));
  const float wP = 1.f / (1.f + __expf(-(red[1] + red[3])));

  const float* rs = kv_cat + bt * KVCAT_;
  const float* rp = rs - (odd ? KVCAT_ : 0);
  for (int c = tid; c < KVCAT_; c += 64) {
    const float v = odd ? (wP * rp[c] + wS * rs[c]) : (wS * rs[c]);
    const unsigned short hv = f32_bf16(v);
    kvh[bt * KVCAT_ + c] = hv;
    if (c < KVR_) kvhT[((long)b * KVR_ + c) * S_ + t] = hv;
  }
}

// ---------------------------------------------------------------------------
// Flash attention: scores = (qh . kvh) * SCALE over the fused 288 dim,
// masked, online softmax, x = attn @ kv_t (256) -> bf16 out.
// Block = 128 threads (4 waves); wave w owns 16 query rows; streams 32-key
// tiles. All fragments are contiguous 16B bf16 loads; P re-laid-out C->A
// through per-wave LDS. Grid: (S/64, H, B).
// ---------------------------------------------------------------------------
__global__ __launch_bounds__(128)
void attn_kernel(const unsigned short* __restrict__ qh,    // (B,S,H,288) bf16
                 const unsigned short* __restrict__ kvh,   // (B,S,288)   bf16
                 const unsigned short* __restrict__ kvhT,  // (B,256,S)   bf16
                 unsigned short* __restrict__ xout)        // (B,S,H,256) bf16
{
  __shared__ __align__(16) unsigned short lp[4][16 * 32];

  const int b = blockIdx.z, h = blockIdx.y;
  const int m0 = blockIdx.x * 64;
  const int wid = threadIdx.x >> 5;
  const int lane = threadIdx.x & 31;
  const int lane16 = lane & 15;
  const int half = lane >> 4;

  const int qrow = m0 + wid * 16 + lane16;           // A-frag row of this lane
  const unsigned short* qbase = qh + (((long)b * S_ + qrow) * H_ + h) * KVCAT_;
  const int rbase = m0 + wid * 16 + 8 * half;        // score-row of element r
  const unsigned short* vbase = kvhT + ((long)b * KVR_ + lane16) * (long)S_;

  float mrow[8], lrow[8];
  v8f acc[16];
#pragma unroll
  for (int r = 0; r < 8; ++r) { mrow[r] = -1e30f; lrow[r] = 0.f; }
#pragma unroll
  for (int cb = 0; cb < 16; ++cb) acc[cb] = v8zero();

  for (int t0 = 0; t0 < S_; t0 += 32) {
    // ---------- scores: two 16x16 tiles over K=288 ----------
    v8f s0 = v8zero(), s1 = v8zero();
    const unsigned short* k0p = kvh + ((long)b * S_ + t0 + lane16) * KVCAT_;
    const unsigned short* k1p = k0p + 16 * KVCAT_;
    __builtin_prefetch(k0p + 32 * KVCAT_, 0, 1);     // next key tile
#pragma unroll
    for (int kc = 0; kc < KVCAT_; kc += 32) {
      Frag16 fa, f0, f1;
      const uint4* pa = (const uint4*)(qbase + kc);
      fa.u4[0] = pa[half]; fa.u4[1] = pa[2 + half];
      const uint4* p0 = (const uint4*)(k0p + kc);
      f0.u4[0] = p0[half]; f0.u4[1] = p0[2 + half];
      const uint4* p1 = (const uint4*)(k1p + kc);
      f1.u4[0] = p1[half]; f1.u4[1] = p1[2 + half];
      s0 = wmma_bf16(fa, f0, s0);
      s1 = wmma_bf16(fa, f1, s1);
    }

    // ---------- mask + online softmax (row reductions over 16 lanes) ------
    const int tc0 = t0 + lane16, tc1 = tc0 + 16;
    float p0[8], p1[8], alpha[8];
#pragma unroll
    for (int r = 0; r < 8; ++r) {
      const int sr = rbase + r;
      // visible = (t<=s && t odd) || t==s
      const bool v0 = ((tc0 <= sr) && (tc0 & 1)) || (tc0 == sr);
      const bool v1 = ((tc1 <= sr) && (tc1 & 1)) || (tc1 == sr);
      float a0 = s0[r] * SCALE_ + (v0 ? 0.f : -1e9f);
      float a1 = s1[r] * SCALE_ + (v1 ? 0.f : -1e9f);
      float mx = fmaxf(a0, a1);
#pragma unroll
      for (int off = 1; off < 16; off <<= 1) mx = fmaxf(mx, __shfl_xor(mx, off, 32));
      const float mnew = fmaxf(mrow[r], mx);
      const float al = __expf(mrow[r] - mnew);
      const float e0 = __expf(a0 - mnew);
      const float e1 = __expf(a1 - mnew);
      float sum = e0 + e1;
#pragma unroll
      for (int off = 1; off < 16; off <<= 1) sum += __shfl_xor(sum, off, 32);
      lrow[r] = lrow[r] * al + sum;
      mrow[r] = mnew;
      alpha[r] = al; p0[r] = e0; p1[r] = e1;
    }
#pragma unroll
    for (int cb = 0; cb < 16; ++cb)
#pragma unroll
      for (int r = 0; r < 8; ++r) acc[cb][r] *= alpha[r];

    // ---------- P: C layout -> A layout through per-wave LDS ----------
    unsigned short* lpw = lp[wid];
#pragma unroll
    for (int r = 0; r < 8; ++r) {
      lpw[(r + 8 * half) * 32 + lane16]      = f32_bf16(p0[r]);
      lpw[(r + 8 * half) * 32 + 16 + lane16] = f32_bf16(p1[r]);
    }
    Frag16 fp;                     // LDS in-order per wave; compiler waits dscnt
    const uint4* pp = (const uint4*)(lpw + lane16 * 32);
    fp.u4[0] = pp[half]; fp.u4[1] = pp[2 + half];

    // ---------- x += P(16x32) @ V(32x256) : 16 WMMAs ----------
#pragma unroll
    for (int cb = 0; cb < 16; ++cb) {
      Frag16 fv;
      const uint4* pv = (const uint4*)(vbase + (long)cb * 16 * S_ + t0);
      fv.u4[0] = pv[half]; fv.u4[1] = pv[2 + half];
      acc[cb] = wmma_bf16(fp, fv, acc[cb]);
    }
  }

  // ---------- epilogue: divide by l, write bf16 x (B,S,H,256) ----------
  unsigned short* xo = xout + (((long)b * S_ + rbase) * H_ + h) * KVR_ + lane16;
#pragma unroll
  for (int cb = 0; cb < 16; ++cb)
#pragma unroll
    for (int r = 0; r < 8; ++r)
      xo[(long)r * (H_ * KVR_) + cb * 16] = f32_bf16(acc[cb][r] / lrow[r]);
}

// ---------------------------------------------------------------------------
// Host orchestration
// ---------------------------------------------------------------------------
extern "C" void kernel_launch(void* const* d_in, const int* in_sizes, int n_in,
                              void* d_out, int out_size, void* d_ws, size_t ws_size,
                              hipStream_t stream)
{
  (void)in_sizes; (void)n_in; (void)out_size; (void)ws_size;

  const float* hidden  = (const float*)d_in[0];   // (B,S,E)
  const float* wq      = (const float*)d_in[1];   // (H*QKD, E)
  const float* wkv_a   = (const float*)d_in[2];   // (KVR+ROPE, E)
  const float* ln_g    = (const float*)d_in[3];
  const float* ln_b    = (const float*)d_in[4];
  const float* wkv_b   = (const float*)d_in[5];   // (H*128, KVR)
  const float* wo      = (const float*)d_in[6];   // (E, H*V)
  const float* fc_c_w  = (const float*)d_in[7];
  const float* fc_c_b  = (const float*)d_in[8];
  const float* fc_p_w  = (const float*)d_in[9];
  const float* fc_p_b  = (const float*)d_in[10];
  float* out = (float*)d_out;                     // (B,S,E) f32

  // ---- workspace carve-up (256B aligned slabs) ----
  size_t off = 0;
  auto grab = [&](size_t bytes) {
    void* p = (char*)d_ws + off;
    off += (bytes + 255) & ~(size_t)255;
    return p;
  };
  const long MBS = (long)B_ * S_;                                  // 4096 rows
  // bf16 operand copies
  unsigned short* hidh  = (unsigned short*)grab(MBS * E_ * 2);
  unsigned short* wqh   = (unsigned short*)grab((long)H_ * QKD_ * E_ * 2);
  unsigned short* wkvah = (unsigned short*)grab((long)KVCAT_ * E_ * 2);
  unsigned short* wkvbh = (unsigned short*)grab((long)H_ * 128 * KVR_ * 2);
  unsigned short* woh   = (unsigned short*)grab((long)E_ * H_ * V_ * 2);
  // pipeline buffers
  unsigned short* qbh   = (unsigned short*)grab(MBS * H_ * QKD_ * 2);   // q proj
  float*          kv_buf= (float*)grab(MBS * KVCAT_ * 4);
  float*          kv_cat= (float*)grab(MBS * KVCAT_ * 4);
  float*          C2buf = (float*)grab((long)S_ * 64 * 4);
  float*          P2buf = (float*)grab((long)S_ * 64 * 4);
  unsigned short* kvh   = (unsigned short*)grab(MBS * KVCAT_ * 2);
  unsigned short* kvhT  = (unsigned short*)grab((long)B_ * KVR_ * S_ * 2);
  unsigned short* qh    = (unsigned short*)grab(MBS * H_ * KVCAT_ * 2); // q_abs||q_pe
  unsigned short* xh    = (unsigned short*)grab(MBS * H_ * KVR_ * 2);
  unsigned short* ohh   = (unsigned short*)grab(MBS * H_ * V_ * 2);

  // 0) one-time bf16 conversion of all GEMM operands
  cvt_f32_bf16_v4<<<2048, 256, 0, stream>>>((const float4*)hidden, (uint2*)hidh,  MBS * E_ / 4);
  cvt_f32_bf16_v4<<<1024, 256, 0, stream>>>((const float4*)wq,     (uint2*)wqh,   (long)H_ * QKD_ * E_ / 4);
  cvt_f32_bf16_v4<<<512,  256, 0, stream>>>((const float4*)wkv_a,  (uint2*)wkvah, (long)KVCAT_ * E_ / 4);
  cvt_f32_bf16_v4<<<512,  256, 0, stream>>>((const float4*)wkv_b,  (uint2*)wkvbh, (long)H_ * 128 * KVR_ / 4);
  cvt_f32_bf16_v4<<<1024, 256, 0, stream>>>((const float4*)wo,     (uint2*)woh,   (long)E_ * H_ * V_ / 4);

  // 1) q = hidden @ wq^T  -> bf16       (M=4096, N=1536, K=2048, NT)
  gemm_kern<1, 1><<<dim3((H_ * QKD_ + GBN - 1) / GBN, MBS / GBM, 1), 256, 0, stream>>>(
      hidh, E_, 0, wqh, E_, 0, qbh, H_ * QKD_, 0, H_ * QKD_, E_);

  // 2) kv = hidden @ wkv_a^T -> f32     (M=4096, N=288, K=2048, NT, masked N)
  gemm_kern<1, 0><<<dim3((KVCAT_ + GBN - 1) / GBN, MBS / GBM, 1), 256, 0, stream>>>(
      hidh, E_, 0, wkvah, E_, 0, kv_buf, KVCAT_, 0, KVCAT_, E_);

  // 3) layernorm(kv_c) || rope(k_pe) -> kv_cat (f32)
  ln_rope_kernel<<<dim3((unsigned)MBS), 256, 0, stream>>>(kv_buf, ln_g, ln_b, kv_cat);

  // 4) rope(q_pe) -> qh[...,256:288]
  rope_q_kernel<<<dim3((unsigned)(MBS * H_ * 16 / 256)), 256, 0, stream>>>(qbh, qh);

  // 5) analytic pos-enc FCs -> C2, P2
  pos_fc_kernel<<<dim3(S_), 64, 0, stream>>>(fc_c_w, fc_c_b, fc_p_w, fc_p_b, C2buf, P2buf);

  // 6) sparse 2x2 gate -> kvh (B,S,288) + kvhT (B,256,S), both bf16
  kv_gate_kernel<<<dim3((unsigned)MBS), 64, 0, stream>>>(C2buf, P2buf, kv_cat, kvh, kvhT);

  // 7) q_abs[h] = q_nope[h] @ wkv_b3[h,:64] -> qh[...,0:256]  (NN, batched)
  gemm_kern<0, 1><<<dim3(KVR_ / GBN, MBS / GBM, H_), 256, 0, stream>>>(
      qbh, H_ * QKD_, QKD_,
      wkvbh, KVR_, (long)128 * KVR_,
      qh, H_ * KVCAT_, KVCAT_,
      KVR_, NOPE_);

  // 8) flash attention -> xh (B,S,H,256) bf16
  attn_kernel<<<dim3(S_ / 64, H_, B_), 128, 0, stream>>>(qh, kvh, kvhT, xh);

  // 9) head value proj: oh[h] = x[h] @ wkv_b3[h,64:128]^T -> bf16 (NT, batched)
  gemm_kern<1, 1><<<dim3(V_ / GBN, MBS / GBM, H_), 256, 0, stream>>>(
      xh, H_ * KVR_, KVR_,
      wkvbh + (long)NOPE_ * KVR_, KVR_, (long)128 * KVR_,
      ohh, H_ * V_, V_,
      V_, KVR_);

  // 10) out = oh @ wo^T -> f32          (M=4096, N=2048, K=1024, NT)
  gemm_kern<1, 0><<<dim3(E_ / GBN, MBS / GBM, 1), 256, 0, stream>>>(
      ohh, H_ * V_, 0, woh, H_ * V_, 0, out, E_, 0, E_, H_ * V_);
}
